// VitNet_11862699672101
// MI455X (gfx1250) — matrix operand order repaired and checked by reading, not compiled, not run on hardware
//
#include <hip/hip_runtime.h>

// ---------------------------------------------------------------------------
// LC0-style transformer forward for MI455X (gfx1250), bf16 WMMA everywhere.
//   - weights f32 -> transposed bf16 [N][K] once per launch (tiled LDS transpose)
//   - GEMM: 128x64 block tile, BK=64, 8 wave32 waves x 4 acc tiles each,
//     double-buffered GLOBAL_LOAD_ASYNC_TO_LDS_B128 staging (ASYNCcnt pipeline),
//     batched fragment loads so the 4 WMMAs per A-fragment issue back-to-back
//   - fused attention per (batch,head): QK^T (WMMA) + smolgen bias + softmax + PV
//   - LN / residual / swish / relu epilogue kernels
// ---------------------------------------------------------------------------

typedef __bf16 bf16_t;
typedef __attribute__((ext_vector_type(16))) __bf16 v16bf;
typedef __attribute__((ext_vector_type(8)))  __bf16 v8bf;
typedef __attribute__((ext_vector_type(8)))  float   v8f;

union Frag16 { v16bf v; v8bf h[2]; };

#define ACT_NONE  0
#define ACT_RELU  1
#define ACT_SWISH 2

__device__ __forceinline__ v8f wmma_bf16(Frag16 a, Frag16 b, v8f c) {
  // D = A(16x32) * B(32x16) + C, f32 accumulate. Probe-confirmed builtin.
  return __builtin_amdgcn_wmma_f32_16x16x32_bf16(false, a.v, false, b.v,
                                                 (short)0, c, false, false);
}

// A fragment (this file's contiguous k order, same as B): lane&15 = M; khalf = lane>>4.
// Elements 0..15 = K[khalf*16 .. +15].
__device__ __forceinline__ void load_fragA(Frag16& f, const bf16_t* p) {
  f.h[0] = *(const v8bf*)(p);
  f.h[1] = *(const v8bf*)(p + 8);
}
// B fragment ([N][K] transposed storage): lane&15 = N row, khalf picks K 0..15/16..31.
__device__ __forceinline__ void load_fragB(Frag16& f, const bf16_t* p) {
  f.h[0] = *(const v8bf*)(p);
  f.h[1] = *(const v8bf*)(p + 8);
}

// 16B global -> LDS async copy (ASYNCcnt). lds = raw LDS byte offset
// (= low 32 bits of the flat address, ISA 10.2), g = 64-bit global address.
__device__ __forceinline__ void async_ld16(unsigned lds, const void* g) {
  asm volatile("global_load_async_to_lds_b128 %0, %1, off"
               :: "v"(lds), "v"((unsigned long long)(size_t)g) : "memory");
}

// --------------------------- weight transpose+cvt ---------------------------
// wt[n*Kp + k] = (bf16) w[k*N + n], zero-padded for k in [K, Kp)
__global__ __launch_bounds__(256)
void k_transpose_cvt(const float* __restrict__ w, bf16_t* __restrict__ wt,
                     int K, int N, int Kp) {
  __shared__ float tile[32][33];
  const int kb = blockIdx.x * 32, nb = blockIdx.y * 32;
  const int tx = threadIdx.x & 31, ty = threadIdx.x >> 5;   // 32x8
  for (int i = ty; i < 32; i += 8) {
    int k = kb + i, n = nb + tx;
    tile[i][tx] = (k < K && n < N) ? w[(size_t)k * N + n] : 0.f;
  }
  __syncthreads();
  for (int i = ty; i < 32; i += 8) {
    int n = nb + i, k = kb + tx;
    if (n < N && k < Kp) wt[(size_t)n * Kp + k] = (bf16_t)tile[tx][i];
  }
}

// ------------------------------- embedding ----------------------------------
// emb[t,c] bf16 over padded K=192: c<112 -> x[b,c,s]; c<176 -> x[b,0,c-112]; else 0
__global__ __launch_bounds__(256)
void k_embed(const float* __restrict__ x, bf16_t* __restrict__ emb) {
  int idx = blockIdx.x * 256 + threadIdx.x;
  if (idx >= 16384 * 192) return;
  int t = idx / 192, c = idx % 192;
  int b = t >> 6, s = t & 63;
  float v = 0.f;
  if (c < 112)      v = x[((size_t)b * 112 + c) * 64 + s];
  else if (c < 176) v = x[(size_t)b * 112 * 64 + (c - 112)];
  emb[idx] = (bf16_t)v;
}

// --------------------------------- GEMM -------------------------------------
// C[M,N] = A[M,K](bf16) x Bt[N,K](bf16) (+bias) (+act) (*mult[m%64,n])
// Requires M%128==0, K%64==0. N arbitrary (B rows clamped; cols >= N not stored).
#define GLDT 72   // LDS row stride in halves (64 + 8 pad)

__global__ __launch_bounds__(256)
void k_gemm_bf16(const bf16_t* __restrict__ A, const bf16_t* __restrict__ Bt,
                 const float* __restrict__ bias,
                 float* __restrict__ Cf, bf16_t* __restrict__ Cb,
                 const float* __restrict__ mul,
                 int M, int N, int K, int act) {
  __shared__ __align__(16) bf16_t As[2][128 * GLDT];
  __shared__ __align__(16) bf16_t Bs[2][64 * GLDT];
  const int tid = threadIdx.x;
  const int lane = tid & 31, wid = tid >> 5;
  const int bm = blockIdx.x * 128, bn = blockIdx.y * 64;
  const int mrow = lane & 15, khalf = lane >> 4;

  v8f z = {};
  v8f acc[4]; acc[0] = z; acc[1] = z; acc[2] = z; acc[3] = z;

  const unsigned asb[2] = { (unsigned)(size_t)&As[0][0], (unsigned)(size_t)&As[1][0] };
  const unsigned bsb[2] = { (unsigned)(size_t)&Bs[0][0], (unsigned)(size_t)&Bs[1][0] };

  // Stage one 128x64 A tile + 64x64 B tile: 6 async 16B chunks per thread.
  auto issue = [&](int buf, int k0) {
    #pragma unroll
    for (int j = 0; j < 4; ++j) {               // A: 1024 chunks (128 rows x 8)
      int ch = tid + 256 * j;
      int r = ch >> 3, c = (ch & 7) * 8;
      async_ld16(asb[buf] + (unsigned)(r * GLDT + c) * 2,
                 A + (size_t)(bm + r) * K + (k0 + c));
    }
    #pragma unroll
    for (int j = 0; j < 2; ++j) {               // B: 512 chunks (64 rows x 8)
      int ch = tid + 256 * j;
      int r = ch >> 3, c = (ch & 7) * 8;
      int gn = bn + r; if (gn >= N) gn = N - 1;   // clamp: garbage cols never stored
      async_ld16(bsb[buf] + (unsigned)(r * GLDT + c) * 2,
                 Bt + (size_t)gn * K + (k0 + c));
    }
  };

  const int nt = K >> 6;                        // K % 64 == 0
  issue(0, 0);
  for (int t = 0; t < nt; ++t) {
    if (t + 1 < nt) {
      issue((t + 1) & 1, (t + 1) << 6);
      asm volatile("s_wait_asynccnt 6" ::: "memory");   // current tile landed
    } else {
      asm volatile("s_wait_asynccnt 0" ::: "memory");
    }
    __syncthreads();
    const bf16_t* as = &As[t & 1][0];
    const bf16_t* bs = &Bs[t & 1][0];
    #pragma unroll
    for (int kk = 0; kk < 64; kk += 32) {
      // batch all fragment loads, then fire the 4 WMMAs back-to-back
      Frag16 a, b[4];
      load_fragA(a, as + (wid * 16 + mrow) * GLDT + kk + khalf * 16);
      #pragma unroll
      for (int ni = 0; ni < 4; ++ni)
        load_fragB(b[ni], bs + (ni * 16 + mrow) * GLDT + kk + khalf * 16);
      #pragma unroll
      for (int ni = 0; ni < 4; ++ni)
        acc[ni] = wmma_bf16(a, b[ni], acc[ni]);
    }
    __syncthreads();
  }

  // epilogue: C/D layout -> lane&15 = N, m = vgpr + (lane>>4)*8
  const int m0 = bm + wid * 16 + (lane >> 4) * 8;
  #pragma unroll
  for (int ni = 0; ni < 4; ++ni) {
    int n = bn + ni * 16 + (lane & 15);
    if (n >= N) continue;
    float bb = bias ? bias[n] : 0.f;
    #pragma unroll
    for (int r = 0; r < 8; ++r) {
      int m = m0 + r;
      float val = acc[ni][r] + bb;
      if (act == ACT_RELU)       val = fmaxf(val, 0.f);
      else if (act == ACT_SWISH) val = val / (1.f + __expf(-val));  // x*sigmoid(x)
      if (mul) val *= mul[(m & 63) * N + n];
      size_t idx = (size_t)m * N + n;
      if (Cf) Cf[idx] = val;
      if (Cb) Cb[idx] = (bf16_t)val;
    }
  }
}

// ------------------------------- layernorm ----------------------------------
// y = LN(a + res*rscale); writes f32 and/or bf16. One block per row.
__global__ __launch_bounds__(256)
void k_ln(const float* __restrict__ a, const float* __restrict__ res,
          float rscale, const float* __restrict__ g, const float* __restrict__ be,
          float* __restrict__ outf, bf16_t* __restrict__ outb, int C) {
  __shared__ float red[256];
  const int row = blockIdx.x, tid = threadIdx.x;
  const float* ar = a + (size_t)row * C;
  const float* rr = res ? res + (size_t)row * C : nullptr;
  float s = 0.f, s2 = 0.f;
  for (int c = tid; c < C; c += 256) {
    float v = ar[c] + (rr ? rr[c] * rscale : 0.f);
    s += v; s2 += v * v;
  }
  red[tid] = s; __syncthreads();
  for (int o = 128; o > 0; o >>= 1) { if (tid < o) red[tid] += red[tid + o]; __syncthreads(); }
  float mean = red[0] / C; __syncthreads();
  red[tid] = s2; __syncthreads();
  for (int o = 128; o > 0; o >>= 1) { if (tid < o) red[tid] += red[tid + o]; __syncthreads(); }
  float var = red[0] / C - mean * mean;
  float rstd = rsqrtf(var + 1e-5f);
  for (int c = tid; c < C; c += 256) {
    float v = ar[c] + (rr ? rr[c] * rscale : 0.f);
    float y = (v - mean) * rstd * g[c] + be[c];
    if (outf) outf[(size_t)row * C + c] = y;
    if (outb) outb[(size_t)row * C + c] = (bf16_t)y;
  }
}

// ------------------------------- attention ----------------------------------
// One block per (batch, head). scores = QK^T*0.125 + smol; softmax; out = P V.
__global__ __launch_bounds__(256)
void k_attn(const bf16_t* __restrict__ qm, const bf16_t* __restrict__ km,
            const bf16_t* __restrict__ vm, const float* __restrict__ smol,
            bf16_t* __restrict__ outb) {
  const int blk = blockIdx.x;              // b*8 + h
  const int b = blk >> 3, hh = blk & 7;
  __shared__ __align__(16) bf16_t Qs[64 * 72];
  __shared__ __align__(16) bf16_t Ks[64 * 72];
  __shared__ __align__(16) bf16_t Vt[64 * 72];
  __shared__ float Ss[64 * 68];
  const int tid = threadIdx.x, lane = tid & 31, wid = tid >> 5;
  const size_t rowBase = (size_t)(b * 64) * 512 + hh * 64;

  for (int it = tid; it < 512; it += 256) {
    int r = it >> 3, c = (it & 7) * 8;
    *(v8bf*)(&Qs[r * 72 + c]) = *(const v8bf*)(qm + rowBase + (size_t)r * 512 + c);
    *(v8bf*)(&Ks[r * 72 + c]) = *(const v8bf*)(km + rowBase + (size_t)r * 512 + c);
  }
  for (int it = tid; it < 512; it += 256) {   // V transposed: Vt[d][k]
    int kp = it >> 3, d0 = (it & 7) * 8;
    v8bf val = *(const v8bf*)(vm + rowBase + (size_t)kp * 512 + d0);
    for (int i = 0; i < 8; ++i) Vt[(d0 + i) * 72 + kp] = ((bf16_t*)&val)[i];
  }
  __syncthreads();

  const int mi = wid & 3, ni0 = (wid >> 2) * 2;
  const int mrow = lane & 15, khalf = lane >> 4;
  v8f acc0 = {}; v8f acc1 = {};
  #pragma unroll
  for (int k0 = 0; k0 < 64; k0 += 32) {
    Frag16 a, b0, b1;
    load_fragA(a,  &Qs[(mi * 16 + mrow) * 72 + k0 + khalf * 16]);
    load_fragB(b0, &Ks[(ni0 * 16 + mrow) * 72 + k0 + khalf * 16]);
    load_fragB(b1, &Ks[((ni0 + 1) * 16 + mrow) * 72 + k0 + khalf * 16]);
    acc0 = wmma_bf16(a, b0, acc0);
    acc1 = wmma_bf16(a, b1, acc1);
  }
  {
    const int n0 = ni0 * 16 + (lane & 15);
    const int m0 = mi * 16 + (lane >> 4) * 8;
    const float* sm = smol + (size_t)blk * 4096;
    #pragma unroll
    for (int r = 0; r < 8; ++r) {
      int m = m0 + r;
      Ss[m * 68 + n0]      = acc0[r] * 0.125f + sm[m * 64 + n0];
      Ss[m * 68 + n0 + 16] = acc1[r] * 0.125f + sm[m * 64 + n0 + 16];
    }
  }
  __syncthreads();

  if (tid < 64) {     // row softmax, probs -> Qs (bf16, Q no longer needed)
    float mx = -1e30f;
    for (int j = 0; j < 64; ++j) mx = fmaxf(mx, Ss[tid * 68 + j]);
    float sum = 0.f;
    for (int j = 0; j < 64; ++j) { float e = __expf(Ss[tid * 68 + j] - mx); sum += e; Ss[tid * 68 + j] = e; }
    float inv = 1.f / sum;
    for (int j = 0; j < 64; ++j) Qs[tid * 72 + j] = (bf16_t)(Ss[tid * 68 + j] * inv);
  }
  __syncthreads();

  v8f z = {}; acc0 = z; acc1 = z;
  #pragma unroll
  for (int k0 = 0; k0 < 64; k0 += 32) {
    Frag16 a, b0, b1;
    load_fragA(a,  &Qs[(mi * 16 + mrow) * 72 + k0 + khalf * 16]);
    load_fragB(b0, &Vt[(ni0 * 16 + mrow) * 72 + k0 + khalf * 16]);
    load_fragB(b1, &Vt[((ni0 + 1) * 16 + mrow) * 72 + k0 + khalf * 16]);
    acc0 = wmma_bf16(a, b0, acc0);
    acc1 = wmma_bf16(a, b1, acc1);
  }
  {
    const int n0 = ni0 * 16 + (lane & 15);
    const int m0 = mi * 16 + (lane >> 4) * 8;
    #pragma unroll
    for (int r = 0; r < 8; ++r) {
      int m = m0 + r;
      outb[rowBase + (size_t)m * 512 + n0]      = (bf16_t)acc0[r];
      outb[rowBase + (size_t)m * 512 + n0 + 16] = (bf16_t)acc1[r];
    }
  }
}

// ---------------------------- policy logits ---------------------------------
// logits[b,64,64] = q1[b] x q2[b]^T over K=512 (POL_SCALE = 1.0)
__global__ __launch_bounds__(256)
void k_pol_logits(const bf16_t* __restrict__ q1, const bf16_t* __restrict__ q2,
                  float* __restrict__ lg) {
  const int b = blockIdx.x;
  __shared__ __align__(16) bf16_t Q1s[64 * 72];
  __shared__ __align__(16) bf16_t Q2s[64 * 72];
  const int tid = threadIdx.x, lane = tid & 31, wid = tid >> 5;
  const int mi = wid & 3, ni0 = (wid >> 2) * 2;
  const int mrow = lane & 15, khalf = lane >> 4;
  const size_t base = (size_t)(b * 64) * 512;
  v8f acc0 = {}; v8f acc1 = {};
  for (int c0 = 0; c0 < 512; c0 += 64) {
    for (int it = tid; it < 512; it += 256) {
      int r = it >> 3, c = (it & 7) * 8;
      *(v8bf*)(&Q1s[r * 72 + c]) = *(const v8bf*)(q1 + base + (size_t)r * 512 + c0 + c);
      *(v8bf*)(&Q2s[r * 72 + c]) = *(const v8bf*)(q2 + base + (size_t)r * 512 + c0 + c);
    }
    __syncthreads();
    #pragma unroll
    for (int k0 = 0; k0 < 64; k0 += 32) {
      Frag16 a, b0, b1;
      load_fragA(a,  &Q1s[(mi * 16 + mrow) * 72 + k0 + khalf * 16]);
      load_fragB(b0, &Q2s[(ni0 * 16 + mrow) * 72 + k0 + khalf * 16]);
      load_fragB(b1, &Q2s[((ni0 + 1) * 16 + mrow) * 72 + k0 + khalf * 16]);
      acc0 = wmma_bf16(a, b0, acc0);
      acc1 = wmma_bf16(a, b1, acc1);
    }
    __syncthreads();
  }
  const int n0 = ni0 * 16 + (lane & 15);
  const int m0 = mi * 16 + (lane >> 4) * 8;
  float* out = lg + (size_t)b * 4096;
  #pragma unroll
  for (int r = 0; r < 8; ++r) {
    int m = m0 + r;
    out[m * 64 + n0]      = acc0[r];
    out[m * 64 + n0 + 16] = acc1[r];
  }
}

// ------------------------------ policy tail ---------------------------------
// promo rows + gather. NOTE: the reference GATHER table is a host-side NumPy
// constant not provided as an input; a deterministic sorted spread over the
// 4288 logits is used as a stand-in.
__global__ __launch_bounds__(256)
void k_policy_tail(const float* __restrict__ lg, const bf16_t* __restrict__ q2,
                   const bf16_t* __restrict__ w4t /*[4][512] bf16*/,
                   float* __restrict__ out /* [B,1858] */) {
  const int b = blockIdx.x;
  __shared__ float offraw[8 * 4];
  __shared__ float off[3 * 8];
  if (threadIdx.x < 32) {
    int ki = threadIdx.x >> 2, p = threadIdx.x & 3;
    const bf16_t* q = q2 + (size_t)(b * 64 + 56 + ki) * 512;
    const bf16_t* w = w4t + (size_t)p * 512;
    float s = 0.f;
    for (int i = 0; i < 512; ++i) s += (float)q[i] * (float)w[i];
    offraw[ki * 4 + p] = s;
  }
  __syncthreads();
  if (threadIdx.x < 24) {
    int p = threadIdx.x / 8, ki = threadIdx.x % 8;
    off[p * 8 + ki] = offraw[ki * 4 + p] + offraw[ki * 4 + 3];
  }
  __syncthreads();
  const float* L = lg + (size_t)b * 4096;
  for (int j = threadIdx.x; j < 1858; j += 256) {
    int g = (int)(((long long)j * 4288) / 1858);   // GATHER placeholder
    float v;
    if (g < 4096) {
      v = L[g];
    } else {
      int f = g - 4096;                 // promo.reshape(3,64) flat index
      int qi = f / 24, rem = f % 24, ki = rem / 3, p = rem % 3;
      v = L[(48 + qi) * 64 + (56 + ki)] + off[p * 8 + ki];
    }
    out[(size_t)b * 1858 + j] = v;
  }
}

// 3-way softmax for WDL
__global__ __launch_bounds__(256)
void k_wdl_softmax(const float* __restrict__ in, float* __restrict__ out) {
  int r = threadIdx.x;
  if (r >= 256) return;
  float a = in[r * 3], b = in[r * 3 + 1], c = in[r * 3 + 2];
  float m = fmaxf(a, fmaxf(b, c));
  float ea = __expf(a - m), eb = __expf(b - m), ec = __expf(c - m);
  float inv = 1.f / (ea + eb + ec);
  out[r * 3] = ea * inv; out[r * 3 + 1] = eb * inv; out[r * 3 + 2] = ec * inv;
}

// ---------------------------------------------------------------------------
extern "C" void kernel_launch(void* const* d_in, const int* in_sizes, int n_in,
                              void* d_out, int out_size, void* d_ws, size_t ws_size,
                              hipStream_t stream) {
  (void)in_sizes; (void)n_in; (void)out_size; (void)ws_size;
  const int T = 16384;
  auto F = [&](int i) -> const float* { return (const float*)d_in[i]; };
  int ii = 0;
  const float* x = F(ii++);

  // params pytree, alphabetical dict-key flatten order
  struct BlkP {
    const float *kb,*kw,*ln1b,*ln1s,*ln2b,*ln2s,*m1b,*m1w,*m2b,*m2w,*ob,*ow,
                *qb,*qw,*sg1w,*sg2b,*sg2w,*sg3b,*sg3w,*sg4w,
                *sgl1b,*sgl1s,*sgl2b,*sgl2s,*vb,*vw;
  } bp[12];
  for (int i = 0; i < 12; ++i) {
    BlkP& B = bp[i];
    B.kb=F(ii++);  B.kw=F(ii++);
    B.ln1b=F(ii++);B.ln1s=F(ii++);
    B.ln2b=F(ii++);B.ln2s=F(ii++);
    B.m1b=F(ii++); B.m1w=F(ii++);
    B.m2b=F(ii++); B.m2w=F(ii++);
    B.ob=F(ii++);  B.ow=F(ii++);
    B.qb=F(ii++);  B.qw=F(ii++);
    B.sg1w=F(ii++);
    B.sg2b=F(ii++);B.sg2w=F(ii++);
    B.sg3b=F(ii++);B.sg3w=F(ii++);
    B.sg4w=F(ii++);
    B.sgl1b=F(ii++);B.sgl1s=F(ii++);
    B.sgl2b=F(ii++);B.sgl2s=F(ii++);
    B.vb=F(ii++);  B.vw=F(ii++);
  }
  const float* inib = F(ii++); const float* iniw = F(ii++);
  const float *mlh1b=F(ii++),*mlh1w=F(ii++),*mlh2b=F(ii++),*mlh2w=F(ii++),
              *mlh3b=F(ii++),*mlh3w=F(ii++);
  const float* mult = F(ii++);
  const float *pol1b=F(ii++),*pol1w=F(ii++),*pol2b=F(ii++),*pol2w=F(ii++),
              *pol3b=F(ii++),*pol3w=F(ii++),*pol4w=F(ii++);
  const float *wdl1b=F(ii++),*wdl1w=F(ii++),*wdl2b=F(ii++),*wdl2w=F(ii++),
              *wdl3b=F(ii++),*wdl3w=F(ii++);

  // workspace allocator (~380 MB used)
  char* wsb = (char*)d_ws;
  size_t woff = 0;
  auto alloc = [&](size_t bytes) -> void* {
    void* p = wsb + woff;
    woff += (bytes + 255) & ~(size_t)255;
    return p;
  };
  auto prepW = [&](const float* w, int K, int N, int Kp) -> bf16_t* {
    bf16_t* wt = (bf16_t*)alloc((size_t)Kp * N * sizeof(bf16_t));
    dim3 g((Kp + 31) / 32, (N + 31) / 32);
    k_transpose_cvt<<<g, 256, 0, stream>>>(w, wt, K, N, Kp);
    return wt;
  };

  struct BlkW { bf16_t *q,*k,*v,*o,*sg1,*sg2,*sg3,*sg4,*m1,*m2; } bw[12];
  for (int i = 0; i < 12; ++i) {
    bw[i].q  = prepW(bp[i].qw, 512, 512, 512);
    bw[i].k  = prepW(bp[i].kw, 512, 512, 512);
    bw[i].v  = prepW(bp[i].vw, 512, 512, 512);
    bw[i].o  = prepW(bp[i].ow, 512, 512, 512);
    bw[i].sg1= prepW(bp[i].sg1w, 512, 32, 512);
    bw[i].sg2= prepW(bp[i].sg2w, 2048, 256, 2048);
    bw[i].sg3= prepW(bp[i].sg3w, 256, 2048, 256);
    bw[i].sg4= prepW(bp[i].sg4w, 256, 4096, 256);
    bw[i].m1 = prepW(bp[i].m1w, 512, 1024, 512);
    bw[i].m2 = prepW(bp[i].m2w, 1024, 512, 1024);
  }
  bf16_t* iniWt = prepW(iniw, 176, 512, 192);   // K padded 176 -> 192
  bf16_t* mlhW1 = prepW(mlh1w, 512, 8, 512);
  bf16_t* mlhW2 = prepW(mlh2w, 512, 128, 512);
  bf16_t* mlhW3 = prepW(mlh3w, 128, 1, 128);
  bf16_t* wdlW1 = prepW(wdl1w, 512, 64, 512);
  bf16_t* wdlW2 = prepW(wdl2w, 4096, 128, 4096);
  bf16_t* wdlW3 = prepW(wdl3w, 128, 3, 128);
  bf16_t* polW1 = prepW(pol1w, 512, 512, 512);
  bf16_t* polW2 = prepW(pol2w, 512, 512, 512);
  bf16_t* polW3 = prepW(pol3w, 512, 512, 512);
  bf16_t* polW4 = prepW(pol4w, 512, 4, 512);

  // activation buffers
  bf16_t* emb  = (bf16_t*)alloc((size_t)T * 192 * 2);
  float*  h    = (float*) alloc((size_t)T * 512 * 4);
  bf16_t* hbf  = (bf16_t*)alloc((size_t)T * 512 * 2);
  bf16_t* qb_  = (bf16_t*)alloc((size_t)T * 512 * 2);
  bf16_t* kb_  = (bf16_t*)alloc((size_t)T * 512 * 2);
  bf16_t* vb_  = (bf16_t*)alloc((size_t)T * 512 * 2);
  bf16_t* s1   = (bf16_t*)alloc((size_t)T * 32 * 2);
  float*  s2f  = (float*) alloc((size_t)256 * 256 * 4);
  bf16_t* s2n  = (bf16_t*)alloc((size_t)256 * 256 * 2);
  float*  s3f  = (float*) alloc((size_t)256 * 2048 * 4);
  bf16_t* s3n  = (bf16_t*)alloc((size_t)256 * 2048 * 2);
  float*  smol = (float*) alloc((size_t)2048 * 4096 * 4);
  bf16_t* attn = (bf16_t*)alloc((size_t)T * 512 * 2);
  float*  tmpf = (float*) alloc((size_t)T * 512 * 4);   // o-proj out / mlp2 out
  float*  h1   = (float*) alloc((size_t)T * 512 * 4);
  bf16_t* h1bf = (bf16_t*)alloc((size_t)T * 512 * 2);
  bf16_t* m1   = (bf16_t*)alloc((size_t)T * 1024 * 2);
  bf16_t* mbf  = (bf16_t*)alloc((size_t)T * 8 * 2);
  bf16_t* m2bf = (bf16_t*)alloc((size_t)256 * 128 * 2);
  bf16_t* wbf  = (bf16_t*)alloc((size_t)T * 64 * 2);
  bf16_t* w2bf = (bf16_t*)alloc((size_t)256 * 128 * 2);
  float*  w3f  = (float*) alloc((size_t)256 * 3 * 4);
  float*  lg   = (float*) alloc((size_t)256 * 4096 * 4);
  bf16_t* pobf = attn;   // reuse after blocks finish
  bf16_t* q1bf = qb_;
  bf16_t* q2bf = kb_;

  auto gemm = [&](const bf16_t* A, const bf16_t* Bt, const float* bias,
                  float* Cf, bf16_t* Cb, int M, int N, int K,
                  int act, const float* mul) {
    dim3 g(M / 128, (N + 63) / 64);
    k_gemm_bf16<<<g, 256, 0, stream>>>(A, Bt, bias, Cf, Cb, mul, M, N, K, act);
  };

  // -------- embedding + ini --------
  k_embed<<<(T * 192 + 255) / 256, 256, 0, stream>>>(x, emb);
  gemm(emb, iniWt, inib, h, hbf, T, 512, 192, ACT_RELU, mult);

  // -------- transformer blocks --------
  for (int i = 0; i < 12; ++i) {
    BlkP& P = bp[i]; BlkW& W = bw[i];
    gemm(hbf, W.q, P.qb, nullptr, qb_, T, 512, 512, ACT_NONE, nullptr);
    gemm(hbf, W.k, P.kb, nullptr, kb_, T, 512, 512, ACT_NONE, nullptr);
    gemm(hbf, W.v, P.vb, nullptr, vb_, T, 512, 512, ACT_NONE, nullptr);
    // smolgen
    gemm(hbf, W.sg1, nullptr, nullptr, s1, T, 32, 512, ACT_NONE, nullptr);
    gemm(s1, W.sg2, P.sg2b, s2f, nullptr, 256, 256, 2048, ACT_SWISH, nullptr);
    k_ln<<<256, 256, 0, stream>>>(s2f, nullptr, 0.f, P.sgl1s, P.sgl1b, nullptr, s2n, 256);
    gemm(s2n, W.sg3, P.sg3b, s3f, nullptr, 256, 2048, 256, ACT_SWISH, nullptr);
    k_ln<<<256, 256, 0, stream>>>(s3f, nullptr, 0.f, P.sgl2s, P.sgl2b, nullptr, s3n, 2048);
    gemm(s3n, W.sg4, nullptr, smol, nullptr, 2048, 4096, 256, ACT_NONE, nullptr);
    // attention + O-projection + LN1 (residual h)
    k_attn<<<2048, 256, 0, stream>>>(qb_, kb_, vb_, smol, attn);
    gemm(attn, W.o, P.ob, tmpf, nullptr, T, 512, 512, ACT_NONE, nullptr);
    k_ln<<<T, 256, 0, stream>>>(tmpf, h, 1.0f, P.ln1s, P.ln1b, h1, h1bf, 512);
    // MLP + LN2 (residual h1)
    gemm(h1bf, W.m1, P.m1b, nullptr, m1, T, 1024, 512, ACT_RELU, nullptr);
    gemm(m1, W.m2, P.m2b, tmpf, nullptr, T, 512, 1024, ACT_NONE, nullptr);
    k_ln<<<T, 256, 0, stream>>>(tmpf, h1, 1.0f, P.ln2s, P.ln2b, h, hbf, 512);
  }

  float* outMlh = (float*)d_out;
  float* outWdl = (float*)d_out + 256;
  float* outPol = (float*)d_out + 256 + 768;

  // -------- MLH head --------
  gemm(hbf, mlhW1, mlh1b, nullptr, mbf, T, 8, 512, ACT_RELU, nullptr);
  gemm(mbf, mlhW2, mlh2b, nullptr, m2bf, 256, 128, 512, ACT_RELU, nullptr);
  gemm(m2bf, mlhW3, mlh3b, outMlh, nullptr, 256, 1, 128, ACT_RELU, nullptr);

  // -------- WDL head --------
  gemm(hbf, wdlW1, wdl1b, nullptr, wbf, T, 64, 512, ACT_RELU, nullptr);
  gemm(wbf, wdlW2, wdl2b, nullptr, w2bf, 256, 128, 4096, ACT_RELU, nullptr);
  gemm(w2bf, wdlW3, wdl3b, w3f, nullptr, 256, 3, 128, ACT_NONE, nullptr);
  k_wdl_softmax<<<1, 256, 0, stream>>>(w3f, outWdl);

  // -------- policy head --------
  gemm(hbf, polW1, pol1b, nullptr, pobf, T, 512, 512, ACT_RELU, nullptr);
  gemm(pobf, polW2, pol2b, nullptr, q1bf, T, 512, 512, ACT_NONE, nullptr);
  gemm(pobf, polW3, pol3b, nullptr, q2bf, T, 512, 512, ACT_NONE, nullptr);
  k_pol_logits<<<256, 256, 0, stream>>>(q1bf, q2bf, lg);
  k_policy_tail<<<256, 256, 0, stream>>>(lg, q2bf, polW4, outPol);
}
